// LearnableUpsamplingLayer_1202590843018
// MI455X (gfx1250) — matrix operand order repaired and checked by reading, not compiled, not run on hardware
//
#include <hip/hip_runtime.h>
#include <hip/hip_bf16.h>

// ---------------------------------------------------------------------------
// Problem constants (from the reference)
// ---------------------------------------------------------------------------
#define BB   8
#define TT   4096
#define TWOT 8192          // 2*T
#define WW   8
#define CC   128
#define FF   128           // NUM_FILTERS
#define KH   5             // FILTER_SIZE
#define KTOT 640           // KH*CC
#define NKB  20            // K blocks of 32 (640/32)
#define NNB  8             // N blocks of 16 (128/16)
#define SLOPE 0.2f

typedef __attribute__((ext_vector_type(16))) __bf16 v16bf;
typedef __attribute__((ext_vector_type(8)))  __bf16 v8bf;
typedef __attribute__((ext_vector_type(8)))  float  v8f;

// LDS y-tile: 20 rows (h0-2 .. h0+17) x 8 wd x 128 c, padded to 136 elems/row
#define YSTRIDE 136
#define YT_ELEMS (20 * 8 * YSTRIDE)                 // 21760 bf16 = 43520 B
#define YT_BYTES (YT_ELEMS * 2)
#define BP_ELEMS (NKB * NNB * 32 * 16)              // 81920 bf16 = 163840 B
#define BP_BYTES (BP_ELEMS * 2)
#define SMEM_BYTES (YT_BYTES + BP_BYTES)            // 207360 B (<320KB WGP LDS)

// ---------------------------------------------------------------------------
// Prep kernel: pack conv weights [640,128] f32 -> bf16 fragments in the exact
// 32x16 bf16 B-matrix lane layout (lanes 0-15: K=0..7,16..23, lanes 16-31: K=8..15,24..31,
// col = lane%16), fragment-major:  frag(kb,nb) -> 32 lanes x 16 bf16 contiguous.
// Also precompute sigmoid(interp_weights).
// ---------------------------------------------------------------------------
__global__ __launch_bounds__(256) void pack_weights_kernel(
    const float* __restrict__ kern,   // [640,128]
    const float* __restrict__ iw,     // [128]
    __bf16* __restrict__ Bpack,       // [20*8*32*16]
    float* __restrict__ sig)          // [128]
{
    const int tid = blockIdx.x * 256 + threadIdx.x;
    if (tid < CC) {
        sig[tid] = 1.0f / (1.0f + __expf(-iw[tid]));
    }
    if (tid < BP_ELEMS) {
        const int e    = tid & 15;          // element within lane chunk
        const int lane = (tid >> 4) & 31;
        const int nb   = (tid >> 9) & 7;
        const int kb   = tid >> 12;
        const int half = lane >> 4;
        const int n    = nb * 16 + (lane & 15);
        const int k    = kb * 32 + 8 * half + (e & 7) + ((e >> 3) << 4);   // B halves: K 8h+0..7 then 16+8h+0..7
        Bpack[tid] = (__bf16)kern[k * FF + n];
    }
}

// ---------------------------------------------------------------------------
// Main fused kernel:
//   Phase 0: async-copy packed B (160KB) global->LDS (ASYNCcnt engine),
//            overlapped with
//   Phase 1: build interleaved y tile (interp + interleave) in LDS as bf16,
//   Phase 2: bf16 WMMA GEMM (M=128, N=128, K=640) entirely from LDS,
//   Phase 3: bias + leaky-relu epilogue, f32 stores.
// Grid: B * (2T/16) = 4096 workgroups of 256 threads (8 waves).
// ---------------------------------------------------------------------------
__global__ __launch_bounds__(256) void upsample_conv_kernel(
    const float*  __restrict__ x,      // [B,T,W,C]
    const float*  __restrict__ bias,   // [F]
    const __bf16* __restrict__ Bpack,  // packed weights (global)
    const float*  __restrict__ sig,    // sigmoid(interp_weights) [C]
    float* __restrict__ out)           // [B,2T,W,F]
{
    extern __shared__ char smem[];
    __bf16* yt = (__bf16*)smem;                    // [20*8][YSTRIDE]
    __bf16* Bs = (__bf16*)(smem + YT_BYTES);       // packed B, LDS copy

    const int b   = blockIdx.x >> 9;               // 512 h-tiles per batch
    const int th  = blockIdx.x & 511;
    const int h0  = th << 4;                       // first output h of this tile
    const int tid = threadIdx.x;

    // ---- Phase 0: kick off async global->LDS copy of packed weights -------
    // 163840 B / 16 B = 10240 chunks; 40 per thread. VDST = WG-relative LDS
    // byte address (hardware adds LDS_BASE); GVS mode: SGPR base + VGPR off.
    {
        const unsigned bs_lds = (unsigned)(size_t)(void*)Bs;
        for (int i = tid; i < BP_BYTES / 16; i += 256) {
            const unsigned ldsa = bs_lds + ((unsigned)i << 4);
            const unsigned goff = (unsigned)i << 4;
            asm volatile("global_load_async_to_lds_b128 %0, %1, %2"
                         :: "v"(ldsa), "v"(goff), "s"(Bpack)
                         : "memory");
        }
    }

    // ---- Phase 1: build interleaved y tile (rows h0-2 .. h0+17) in LDS ----
    for (int idx = tid; idx < 20 * 8 * 128; idx += 256) {
        const int c  = idx & 127;
        const int wd = (idx >> 7) & 7;
        const int hl = idx >> 10;            // 0..19
        const int h  = h0 - 2 + hl;
        float v = 0.0f;
        if (h >= 0 && h < TWOT) {
            const int t = h >> 1;
            const long base = (((long)b * TT + t) * WW + wd) * CC + c;
            if (h & 1) {                     // interpolated row
                const float w  = sig[c];
                const float x0 = x[base];
                const float x1 = (t + 1 < TT) ? x[base + (long)WW * CC] : 0.0f;
                v = w * x0 + (1.0f - w) * x1;
            } else {                         // original row
                v = x[base];
            }
        }
        yt[(hl * 8 + wd) * YSTRIDE + c] = (__bf16)v;
    }

    // All async B chunks landed + all yt writes visible to every wave.
    asm volatile("s_wait_asynccnt 0" ::: "memory");
    __syncthreads();

    // ---- Phase 2: WMMA GEMM, one 16x128 strip per wave, all-LDS operands --
    const int wave = tid >> 5;               // 0..7
    const int lane = tid & 31;
    const int half = lane >> 4;              // 0/1
    const int l16  = lane & 15;

    // This lane's A row within the wave's 16-row tile (A layout: M = lane%16)
    const int wd_a = l16 & 7;
    const int dh_a = 2 * wave + (l16 >> 3);  // local h row (before kh shift)

    v8f acc[NNB];
#pragma unroll
    for (int nb = 0; nb < NNB; ++nb) acc[nb] = (v8f){0.f,0.f,0.f,0.f,0.f,0.f,0.f,0.f};

    for (int kb = 0; kb < NKB; ++kb) {
        const int kh = kb >> 2;              // which conv tap
        const int c0 = (kb & 3) << 5;        // c offset of this 32-wide K block
        const int abase = ((dh_a + kh) * 8 + wd_a) * YSTRIDE + c0 + 8 * half;

        // A fragment: two 16-byte LDS loads (K chunks 0..7 / 16..23 per half)
        const v8bf a_lo = *(const v8bf*)(&yt[abase]);
        const v8bf a_hi = *(const v8bf*)(&yt[abase + 16]);
        const v16bf afrag = __builtin_shufflevector(a_lo, a_hi,
            0, 1, 2, 3, 4, 5, 6, 7, 8, 9, 10, 11, 12, 13, 14, 15);

#pragma unroll
        for (int nb = 0; nb < NNB; ++nb) {
            // B fragment: 32 contiguous bytes per lane from LDS
            const v16bf bfrag =
                *(const v16bf*)(Bs + (((kb * NNB + nb) * 32 + lane) << 4));
            acc[nb] = __builtin_amdgcn_wmma_f32_16x16x32_bf16(
                false, afrag, false, bfrag, (short)0, acc[nb], false, false);
        }
    }

    // ---- Phase 3: bias + leaky relu epilogue, direct f32 stores -----------
    // D layout: col N = lane%16 ; VGPR r -> row M = r + 8*(lane/16)
#pragma unroll
    for (int nb = 0; nb < NNB; ++nb) {
        const int f = nb * 16 + l16;
        const float bv = bias[f];
#pragma unroll
        for (int r = 0; r < 8; ++r) {
            const int m  = r + 8 * half;
            const int gr = 16 * wave + m;            // row in 128-row tile
            const int h  = h0 + (gr >> 3);
            const int wd = gr & 7;
            float v = acc[nb][r] + bv;
            v = (v > 0.0f) ? v : SLOPE * v;
            out[(((long)b * TWOT + h) * WW + wd) * FF + f] = v;
        }
    }
}

// ---------------------------------------------------------------------------
// Launcher
// ---------------------------------------------------------------------------
extern "C" void kernel_launch(void* const* d_in, const int* in_sizes, int n_in,
                              void* d_out, int out_size, void* d_ws, size_t ws_size,
                              hipStream_t stream) {
    const float* x    = (const float*)d_in[0];   // [8,4096,8,128]
    const float* iw   = (const float*)d_in[1];   // [128]
    const float* kern = (const float*)d_in[2];   // [5,1,128,128]
    const float* bias = (const float*)d_in[3];   // [128]
    float* out = (float*)d_out;

    // Workspace layout: Bpack (160 KB bf16) then sig (128 f32)
    __bf16* Bpack = (__bf16*)d_ws;
    float*  sig   = (float*)((char*)d_ws + (size_t)BP_BYTES);

    // 1) pack weights + sigmoid (81920 elems -> 320 blocks of 256)
    pack_weights_kernel<<<dim3(320), dim3(256), 0, stream>>>(kern, iw, Bpack, sig);

    // 2) fused interleave + conv + activation: 8 * 512 = 4096 WGs,
    //    dynamic LDS = y tile (42.5 KB) + packed B (160 KB)
    upsample_conv_kernel<<<dim3(BB * (TWOT / 16)), dim3(256), SMEM_BYTES, stream>>>(
        x, bias, Bpack, sig, out);
}
